// Loss_40338332844270
// MI455X (gfx1250) — matrix operand order repaired and checked by reading, not compiled, not run on hardware
//
#include <hip/hip_runtime.h>
#include <hip/hip_fp16.h>
#include <math.h>

typedef __attribute__((ext_vector_type(16))) _Float16 v16h;
typedef __attribute__((ext_vector_type(8)))  float    v8f;

#define B_SZ   16384
#define D_SZ   512
#define NPAIR  8192

// ---- ws layout (in floats) ----
#define WS_INVN   0                 // 3*B_SZ inverse norms
#define WS_NLL    (3*B_SZ)          // 1
#define WS_COXT   (WS_NLL + 1)      // 64 partials (weighted terms)
#define WS_COXE   (WS_COXT + 64)    // 64 partials (event counts)
#define WS_CON    (WS_COXE + 64)    // 64 partials (contrastive sums)

// ---------------------------------------------------------------------------
// Kernel 1: per-row inverse L2 norms of rep1/rep2/rep3 (f32, exact).
// One block per (rep,row). 256 threads, 2 elements each.
// ---------------------------------------------------------------------------
__global__ void k_invnorm(const float* __restrict__ r1,
                          const float* __restrict__ r2,
                          const float* __restrict__ r3,
                          float* __restrict__ invn) {
    __shared__ float red[256];
    int row = blockIdx.x & (B_SZ - 1);
    int rep = blockIdx.x >> 14;
    const float* r = (rep == 0) ? r1 : ((rep == 1) ? r2 : r3);
    const float* p = r + (size_t)row * D_SZ;
    int t = threadIdx.x;
    float a = p[t];
    float b = p[t + 256];
    red[t] = a * a + b * b;
    __syncthreads();
    for (int s = 128; s > 0; s >>= 1) {
        if (t < s) red[t] += red[t + s];
        __syncthreads();
    }
    if (t == 0) {
        float n = sqrtf(red[0]);
        n = fmaxf(n, 1e-8f);            // EPS_COS
        invn[rep * B_SZ + row] = 1.0f / n;
    }
}

// ---------------------------------------------------------------------------
// Kernel 2: NLL = -mean(score[i, event[i]]). Single block, deterministic tree.
// ---------------------------------------------------------------------------
__global__ void k_nll(const float* __restrict__ score,
                      const int* __restrict__ event,
                      float* __restrict__ out_nll) {
    __shared__ float red[1024];
    int t = threadIdx.x;
    float s = 0.0f;
    for (int i = t; i < B_SZ; i += 1024) s += score[2 * i + event[i]];
    red[t] = s;
    __syncthreads();
    for (int st = 512; st > 0; st >>= 1) {
        if (t < st) red[t] += red[t + st];
        __syncthreads();
    }
    if (t == 0) out_nll[0] = -red[0] / (float)B_SZ;
}

// ---------------------------------------------------------------------------
// Kernel 3: Cox term without sorting.
// cum_i = sum_j exp(h_j) * [t_j > t_i  ||  (t_j == t_i && j <= i)]
// term_i = ev_i * (h_i - log(cum_i + 1e-6)).  64 blocks x 256 threads,
// thread owns one i; j loops over LDS-staged chunks in fixed order.
// ---------------------------------------------------------------------------
__global__ void k_cox(const float* __restrict__ hazard,
                      const float* __restrict__ timev,
                      const int* __restrict__ event,
                      float* __restrict__ termP,
                      float* __restrict__ evP) {
    __shared__ float tch[2048];
    __shared__ float ech[2048];
    __shared__ float red[256];
    int t = threadIdx.x;
    int i = blockIdx.x * 256 + t;
    float ti = timev[i];
    float hi = hazard[i];
    float cum = 0.0f;
    for (int c = 0; c < B_SZ; c += 2048) {
        __syncthreads();
        for (int j = t; j < 2048; j += 256) {
            int jj = c + j;
            tch[j] = timev[jj];
            ech[j] = expf(hazard[jj]);
        }
        __syncthreads();
        for (int j = 0; j < 2048; ++j) {
            float tj = tch[j];
            bool inc = (tj > ti) || ((tj == ti) && ((c + j) <= i));
            cum += inc ? ech[j] : 0.0f;
        }
    }
    float ev = (float)event[i];
    float term = ev * (hi - logf(cum + 1e-6f));
    red[t] = term;
    __syncthreads();
    for (int s = 128; s > 0; s >>= 1) { if (t < s) red[t] += red[t + s]; __syncthreads(); }
    if (t == 0) termP[blockIdx.x] = red[0];
    __syncthreads();
    red[t] = ev;
    __syncthreads();
    for (int s = 128; s > 0; s >>= 1) { if (t < s) red[t] += red[t + s]; __syncthreads(); }
    if (t == 0) evP[blockIdx.x] = red[0];
}

// ---------------------------------------------------------------------------
// Kernel 4: contrastive term via WMMA f16->f32.
// One wave = 16 pairs. 9 (u,v) dot groups per pair; diagonal of each 16x16
// WMMA accumulator holds the 16 per-pair dots; scale by f32 inverse norms.
// ---------------------------------------------------------------------------
__device__ __forceinline__ v16h pack16(float4 a, float4 b, float4 c, float4 d) {
    v16h v;
    v[0]  = (_Float16)a.x; v[1]  = (_Float16)a.y; v[2]  = (_Float16)a.z; v[3]  = (_Float16)a.w;
    v[4]  = (_Float16)b.x; v[5]  = (_Float16)b.y; v[6]  = (_Float16)b.z; v[7]  = (_Float16)b.w;
    v[8]  = (_Float16)c.x; v[9]  = (_Float16)c.y; v[10] = (_Float16)c.z; v[11] = (_Float16)c.w;
    v[12] = (_Float16)d.x; v[13] = (_Float16)d.y; v[14] = (_Float16)d.z; v[15] = (_Float16)d.w;
    return v;
}

__global__ void k_con(const float* __restrict__ rep1,
                      const float* __restrict__ rep2,
                      const float* __restrict__ rep3,
                      const int* __restrict__ x1,
                      const int* __restrict__ x2,
                      const float* __restrict__ invn,
                      float* __restrict__ conP) {
    __shared__ float wsum[8];
    int t    = threadIdx.x;
    int lane = t & 31;
    int wv   = t >> 5;
    int tile = blockIdx.x * 8 + wv;          // 64 blocks * 8 waves = 512 tiles
    int r    = lane & 15;                    // row slot 0..15 within tile
    int hi   = lane >> 4;                    // lane half
    int p    = tile * 16 + r;
    int ra   = x1[p];
    int rb   = x2[p];

    const float* RP[3] = { rep1, rep2, rep3 };

    // 9 groups: (u_rep, v_rep, u-is-a-row, v-is-a-row, weight)
    // dis_xx: (r1a,r2a) (r1a,r3a) (r2a,r3a)  w=-0.5  (alpha)
    // dis_yy: (r1b,r2b) (r1b,r3b) (r2b,r3b)  w=-0.5  (beta)
    // dis_xy: (r1a,r1b) (r2a,r2b) (r3a,r3b)  w=+1
    const int   UR[9] = {0, 0, 1, 0, 0, 1, 0, 1, 2};
    const int   VR[9] = {1, 2, 2, 1, 2, 2, 0, 1, 2};
    const int   UA[9] = {1, 1, 1, 0, 0, 0, 1, 1, 1};
    const int   VA[9] = {1, 1, 1, 0, 0, 0, 0, 0, 0};
    const float W [9] = {-0.5f, -0.5f, -0.5f, -0.5f, -0.5f, -0.5f, 1.0f, 1.0f, 1.0f};

    float zacc = 0.0f;

#pragma unroll
    for (int gi = 0; gi < 9; ++gi) {
        int urow = UA[gi] ? ra : rb;
        int vrow = VA[gi] ? ra : rb;
        const float* up = RP[UR[gi]] + (size_t)urow * D_SZ;
        const float* vp = RP[VR[gi]] + (size_t)vrow * D_SZ;

        v8f c = {0.f, 0.f, 0.f, 0.f, 0.f, 0.f, 0.f, 0.f};
        for (int k0 = 0; k0 < D_SZ; k0 += 32) {
            // A fragment (16-bit A 16x32): halves 0-7 = K[k0+8*hi .. +7],
            //                              halves 8-15 = K[k0+16+8*hi .. +7]
            const float* ua = up + k0 + 8 * hi;
            float4 a0 = *(const float4*)(ua);
            float4 a1 = *(const float4*)(ua + 4);
            float4 a2 = *(const float4*)(ua + 16);
            float4 a3 = *(const float4*)(ua + 20);
            // B fragment (16-bit B 32x16): halves 0-7 = K[k0+8*hi .. +7],
            //                              halves 8-15 = K[k0+16+8*hi .. +7]
            const float* vb = vp + k0 + 8 * hi;
            float4 b0 = *(const float4*)(vb);
            float4 b1 = *(const float4*)(vb + 4);
            float4 b2 = *(const float4*)(vb + 16);
            float4 b3 = *(const float4*)(vb + 20);

            v16h af = pack16(a0, a1, a2, a3);
            v16h bf = pack16(b0, b1, b2, b3);
            c = __builtin_amdgcn_wmma_f32_16x16x32_f16(
                    false, af, false, bf, (short)0, c, false, false);
        }

        // diag(i): i<8 -> lane i, vgpr i;  i>=8 -> lane i+16, vgpr i-8.
        // Both cases land on a lane whose (ra,rb) are exactly pair i's rows.
        float invU = invn[UR[gi] * B_SZ + urow];
        float invV = invn[VR[gi] * B_SZ + vrow];
        float sc = W[gi] * invU * invV;
        float dsel = 0.0f;
#pragma unroll
        for (int g = 0; g < 8; ++g) {
            if (lane == g || lane == g + 24) dsel = c[g];
        }
        zacc += sc * dsel;
    }

    bool active = (lane < 8) || (lane >= 24);
    float conv = active ? log1pf(expf(0.2f + zacc)) : 0.0f;   // margin = 0.2
    for (int off = 16; off > 0; off >>= 1)
        conv += __shfl_xor(conv, off, 32);
    if (lane == 0) wsum[wv] = conv;
    __syncthreads();
    if (t == 0) {
        float s = 0.0f;
        for (int w = 0; w < 8; ++w) s += wsum[w];
        conP[blockIdx.x] = s;
    }
}

// ---------------------------------------------------------------------------
// Final: combine partials into the scalar loss.
// ---------------------------------------------------------------------------
__global__ void k_final(const float* __restrict__ ws, float* __restrict__ out) {
    if (threadIdx.x == 0 && blockIdx.x == 0) {
        float nll = ws[WS_NLL];
        float ts = 0.0f, es = 0.0f, cs = 0.0f;
        for (int i = 0; i < 64; ++i) {
            ts += ws[WS_COXT + i];
            es += ws[WS_COXE + i];
            cs += ws[WS_CON + i];
        }
        float cox = -ts / (es + 1e-6f);
        float con = cs / (float)NPAIR;
        out[0] = nll + cox + 0.3f * con;   // trade_off = 0.3
    }
}

extern "C" void kernel_launch(void* const* d_in, const int* in_sizes, int n_in,
                              void* d_out, int out_size, void* d_ws, size_t ws_size,
                              hipStream_t stream) {
    (void)in_sizes; (void)n_in; (void)out_size; (void)ws_size;
    const float* rep1   = (const float*)d_in[0];
    const float* rep2   = (const float*)d_in[1];
    const float* rep3   = (const float*)d_in[2];
    const float* hazard = (const float*)d_in[3];
    const float* score  = (const float*)d_in[4];
    const float* timev  = (const float*)d_in[5];
    const int*   event  = (const int*)d_in[6];
    const int*   x1     = (const int*)d_in[7];
    const int*   x2     = (const int*)d_in[8];
    float* ws  = (float*)d_ws;
    float* out = (float*)d_out;

    k_invnorm<<<dim3(3 * B_SZ), dim3(256), 0, stream>>>(rep1, rep2, rep3, ws + WS_INVN);
    k_nll    <<<dim3(1),        dim3(1024), 0, stream>>>(score, event, ws + WS_NLL);
    k_cox    <<<dim3(64),       dim3(256), 0, stream>>>(hazard, timev, event,
                                                        ws + WS_COXT, ws + WS_COXE);
    k_con    <<<dim3(64),       dim3(256), 0, stream>>>(rep1, rep2, rep3, x1, x2,
                                                        ws + WS_INVN, ws + WS_CON);
    k_final  <<<dim3(1),        dim3(32),  0, stream>>>(ws, out);
}
